// LocalMultiHeadAttention_69664369541437
// MI455X (gfx1250) — matrix operand order repaired and checked by reading, not compiled, not run on hardware
//
#include <hip/hip_runtime.h>

typedef unsigned short u16;
typedef unsigned int   u32;

typedef __attribute__((ext_vector_type(16))) __bf16 v16bf;
typedef __attribute__((ext_vector_type(8)))  float  v8f;
typedef __attribute__((ext_vector_type(4)))  unsigned int u32x4;

union BF16Frag { u32x4 u[2]; v16bf v; };

#define TT 1024          // T
#define DD 1024          // D
#define HH 16            // heads
#define DH 64            // head dim
#define BB 2             // batch
#define MROWS (BB*TT)    // 2048
#define TWO_M 2097152    // B*T*D

__device__ __forceinline__ u16 f2bf(float x) {
    u32 u = __float_as_uint(x);
    u32 r = u + 0x7FFFu + ((u >> 16) & 1u);
    return (u16)(r >> 16);
}
__device__ __forceinline__ u32 pack2(float a, float b) {
    return (u32)f2bf(a) | ((u32)f2bf(b) << 16);
}
__device__ __forceinline__ v8f wmma_bf16(v16bf a, v16bf b, v8f c) {
    return __builtin_amdgcn_wmma_f32_16x16x32_bf16(
        false, a, false, b, (short)0, c, false, false);
}

// ---------------------------------------------------------------------------
// Kernel 1: convert x/context to bf16; swizzle weights + E-band rows into the
// WMMA B-operand fragment layout:
//   frag[(kc*NT + nt)*32 + lane][dword j] = {W[k, n], W[k+1, n]} (bf16 pair)
//   k = kc*32 + (lane&16) + 2j , n = nt*16 + (lane&15)
// ---------------------------------------------------------------------------
#define WFRAG_ELEMS 524288          // 32 kc * 64 nt * 32 lanes * 8 dwords
#define PACK_TOTAL (TWO_M + TWO_M + 4*WFRAG_ELEMS + 2560)

__global__ __launch_bounds__(256) void pack_kernel(
    const float* __restrict__ x, const float* __restrict__ cx,
    const float* __restrict__ lut,
    const float* __restrict__ Wq, const float* __restrict__ Wk,
    const float* __restrict__ Wv, const float* __restrict__ Wo,
    u16* __restrict__ xb, u16* __restrict__ cb,
    u32* __restrict__ wqf, u32* __restrict__ wkf,
    u32* __restrict__ wvf, u32* __restrict__ wof,
    u32* __restrict__ ef)
{
    long long idx = (long long)blockIdx.x * 256 + threadIdx.x;
    if (idx < TWO_M) { xb[idx] = f2bf(x[idx]); return; }
    idx -= TWO_M;
    if (idx < TWO_M) { cb[idx] = f2bf(cx[idx]); return; }
    idx -= TWO_M;
    if (idx < 4 * WFRAG_ELEMS) {
        int w = (int)(idx >> 19);
        int p = (int)(idx & (WFRAG_ELEMS - 1));
        int j    = p & 7;
        int lane = (p >> 3) & 31;
        int nt   = (p >> 8) & 63;
        int kc   = p >> 14;
        int k = kc * 32 + (lane & 16) + 2 * j;
        int n = nt * 16 + (lane & 15);
        const float* W = (w == 0) ? Wq : (w == 1) ? Wk : (w == 2) ? Wv : Wo;
        u32* Wf = (w == 0) ? wqf : (w == 1) ? wkf : (w == 2) ? wvf : wof;
        Wf[p] = pack2(W[(size_t)k * DD + n], W[(size_t)(k + 1) * DD + n]);
        return;
    }
    idx -= 4 * WFRAG_ELEMS;
    if (idx < 2560) {   // E-band frags: 2 kc * 5 dt * 32 lanes * 8 dwords
        int p    = (int)idx;
        int j    = p & 7;
        int lane = (p >> 3) & 31;
        int rest = p >> 8;
        int dt = rest % 5, kc = rest / 5;
        int d  = dt * 16 + (lane & 15);           // d = (j - i) + 32 in [0,79]
        int dh = kc * 32 + (lane & 16) + 2 * j;
        const float* row = lut + (size_t)(991 + d) * DH;   // T-1-32+d
        ef[p] = pack2(row[dh], row[dh + 1]);
    }
}

// ---------------------------------------------------------------------------
// Kernel 2: fused QKV projection GEMM, bf16 WMMA, one wave = 64x64 tile.
// blockIdx.z: 0 -> Q = xb@Wq, 1 -> K = cb@Wk, 2 -> V(transposed) = cb@Wv.
// ---------------------------------------------------------------------------
__global__ __launch_bounds__(128) void gemm_qkv_kernel(
    const u16* __restrict__ xb, const u16* __restrict__ cb,
    const u32* __restrict__ wqf, const u32* __restrict__ wkf,
    const u32* __restrict__ wvf,
    u16* __restrict__ Qw, u16* __restrict__ Kw, u16* __restrict__ Vt)
{
    int mode = blockIdx.z;
    const u16* A  = (mode == 0) ? xb : cb;
    const u32* Wf = (mode == 0) ? wqf : (mode == 1) ? wkf : wvf;

    int tid = threadIdx.x, lane = tid & 31, wave = tid >> 5;
    int m0 = blockIdx.x * 128 + (wave & 1) * 64;
    int n0 = blockIdx.y * 128 + (wave >> 1) * 64;
    int mr = lane & 15, hs = lane >> 4;

    v8f acc[4][4] = {};
    for (int kc = 0; kc < 32; ++kc) {
        BF16Frag a[4], b[4];
#pragma unroll
        for (int mt = 0; mt < 4; ++mt) {
            const u16* p = A + (size_t)(m0 + mt * 16 + mr) * DD + kc * 32 + hs * 16;
            a[mt].u[0] = *(const u32x4*)p;
            a[mt].u[1] = *(const u32x4*)(p + 8);
        }
#pragma unroll
        for (int nt = 0; nt < 4; ++nt) {
            const u32* p = Wf + ((size_t)(kc * 64 + (n0 >> 4) + nt) * 32 + lane) * 8;
            b[nt].u[0] = *(const u32x4*)p;
            b[nt].u[1] = *(const u32x4*)(p + 4);
        }
#pragma unroll
        for (int mt = 0; mt < 4; ++mt)
#pragma unroll
            for (int nt = 0; nt < 4; ++nt)
                acc[mt][nt] = wmma_bf16(a[mt].v, b[nt].v, acc[mt][nt]);
    }
#pragma unroll
    for (int mt = 0; mt < 4; ++mt)
        for (int nt = 0; nt < 4; ++nt)
#pragma unroll
            for (int r = 0; r < 8; ++r) {
                int grow = m0 + mt * 16 + r + hs * 8;   // b*T + t
                int gcol = n0 + nt * 16 + mr;           // h*64 + dc
                int bb = grow >> 10, t = grow & 1023;
                int h = gcol >> 6, dc = gcol & 63;
                u16 val = f2bf(acc[mt][nt][r]);
                if (mode == 2)
                    Vt[((size_t)(bb * HH + h) * DH + dc) * TT + t] = val;       // [B,H,dh,T]
                else {
                    u16* O = (mode == 0) ? Qw : Kw;
                    O[((size_t)(bb * HH + h) * TT + t) * DH + dc] = val;        // [B,H,T,dh]
                }
            }
}

// ---------------------------------------------------------------------------
// Kernel 3: banded attention. One wave per (b, h, 16-query tile).
// Key window for tile i0: j in [i0-32, i0+47] -> 80 columns (5 WMMA N-tiles).
// rel term: S2 = Qtile @ Eband^T, shifted per-row via LDS gather.
// ---------------------------------------------------------------------------
__global__ __launch_bounds__(128) void attn_kernel(
    const u16* __restrict__ Qg, const u16* __restrict__ Kg,
    const u16* __restrict__ Vt, const u32* __restrict__ Ef,
    u16* __restrict__ ctxout)
{
    __shared__ __align__(16) float s2buf[4][16 * 80];
    __shared__ __align__(16) u16   pbuf [4][16 * 96];

    int tid = threadIdx.x, lane = tid & 31, wave = tid >> 5;
    int wg = blockIdx.x * 4 + wave;
    int it = wg & 63, h = (wg >> 6) & 15, b = wg >> 10;
    int bh = b * HH + h;
    int i0 = it * 16;
    int mr = lane & 15, hs = lane >> 4, hs16 = lane & 16, mo = hs * 8;

    const u16* Qp = Qg + (size_t)bh * TT * DH;
    const u16* Kp = Kg + (size_t)bh * TT * DH;
    const u16* Vp = Vt + (size_t)bh * TT * DH;

    // A operand: Q tile rows i0..i0+15, two K-chunks of 32 over dh=64
    BF16Frag aQ[2];
#pragma unroll
    for (int kc = 0; kc < 2; ++kc) {
        const u16* p = Qp + (size_t)(i0 + mr) * DH + kc * 32 + hs * 16;
        aQ[kc].u[0] = *(const u32x4*)p;
        aQ[kc].u[1] = *(const u32x4*)(p + 8);
    }

    // S1 = Q @ K^T over the 80-wide window (clamped loads; masked later)
    v8f s1[5];
#pragma unroll
    for (int jt = 0; jt < 5; ++jt) {
        int j  = i0 - 32 + jt * 16 + mr;
        int jc = j < 0 ? 0 : (j > TT - 1 ? TT - 1 : j);
        v8f acc = {};
#pragma unroll
        for (int kc = 0; kc < 2; ++kc) {
            BF16Frag bk;
            const u16* p = Kp + (size_t)jc * DH + kc * 32 + hs16;
            bk.u[0] = *(const u32x4*)p;
            bk.u[1] = *(const u32x4*)(p + 8);
            acc = wmma_bf16(aQ[kc].v, bk.v, acc);
        }
        s1[jt] = acc;
    }

    // S2 = Q @ Eband^T  (columns d = 0..79, d = j - i + 32)
    v8f s2[5];
#pragma unroll
    for (int dt = 0; dt < 5; ++dt) {
        v8f acc = {};
#pragma unroll
        for (int kc = 0; kc < 2; ++kc) {
            BF16Frag be;
            const u32* p = Ef + ((size_t)(kc * 5 + dt) * 32 + lane) * 8;
            be.u[0] = *(const u32x4*)p;
            be.u[1] = *(const u32x4*)(p + 4);
            acc = wmma_bf16(aQ[kc].v, be.v, acc);
        }
        s2[dt] = acc;
    }

    // stage S2 in LDS for the per-row diagonal shift
    float* S2 = s2buf[wave];
#pragma unroll
    for (int dt = 0; dt < 5; ++dt)
#pragma unroll
        for (int r = 0; r < 8; ++r)
            S2[(r + mo) * 80 + dt * 16 + mr] = s2[dt][r];
    __syncthreads();

    // scores + band mask + softmax (rows live in 16-lane halves)
    float pv[5][8], rinv[8];
#pragma unroll
    for (int r = 0; r < 8; ++r) {
        int   m  = r + mo;
        float mx = -1e30f;
        float scr[5];
#pragma unroll
        for (int jt = 0; jt < 5; ++jt) {
            int c = jt * 16 + mr;
            int j = i0 - 32 + c;
            int d = c - m;                       // j - i + 32
            float s = s1[jt][r];
            if (d >= 0 && d < 80) s += S2[m * 80 + d];
            s *= 0.125f;                         // 1/sqrt(64)
            bool ok = (j >= 0) && (j < TT) && (d >= 0) && (d <= 64);
            scr[jt] = ok ? s : -1e30f;
            mx = fmaxf(mx, scr[jt]);
        }
#pragma unroll
        for (int off = 1; off < 16; off <<= 1)
            mx = fmaxf(mx, __shfl_xor(mx, off, 32));
        float sum = 0.f;
#pragma unroll
        for (int jt = 0; jt < 5; ++jt) {
            pv[jt][r] = __expf(scr[jt] - mx);
            sum += pv[jt][r];
        }
#pragma unroll
        for (int off = 1; off < 16; off <<= 1)
            sum += __shfl_xor(sum, off, 32);
        rinv[r] = 1.0f / sum;
    }

    // stage P (bf16) in LDS, pad cols 80..95 with zeros
    u16* P = pbuf[wave];
#pragma unroll
    for (int jt = 0; jt < 5; ++jt)
#pragma unroll
        for (int r = 0; r < 8; ++r)
            P[(r + mo) * 96 + jt * 16 + mr] = f2bf(pv[jt][r]);
#pragma unroll
    for (int z = 0; z < 8; ++z) {
        int idx = lane * 8 + z;
        P[(idx >> 4) * 96 + 80 + (idx & 15)] = 0;
    }
    __syncthreads();

    // A operand for P (3 K-chunks of 32 over 96), from LDS (ds_load_b128)
    BF16Frag aP[3];
#pragma unroll
    for (int kc = 0; kc < 3; ++kc) {
        const u16* p = P + mr * 96 + kc * 32 + hs * 16;
        aP[kc].u[0] = *(const u32x4*)p;
        aP[kc].u[1] = *(const u32x4*)(p + 8);
    }

    // O = P @ V_window   (V transposed: contiguous along keys)
    v8f o[4] = {};
#pragma unroll
    for (int kc = 0; kc < 3; ++kc) {
        int kstart = i0 - 32 + kc * 32 + hs16;            // multiple of 16
        int ks = (kstart < 0 || kstart > TT - 16) ? 0 : kstart; // OOB rows hit P==0
#pragma unroll
        for (int nt = 0; nt < 4; ++nt) {
            BF16Frag bv;
            const u16* p = Vp + (size_t)(nt * 16 + mr) * TT + ks;
            bv.u[0] = *(const u32x4*)p;
            bv.u[1] = *(const u32x4*)(p + 8);
            o[nt] = wmma_bf16(aP[kc].v, bv.v, o[nt]);
        }
    }

#pragma unroll
    for (int nt = 0; nt < 4; ++nt)
#pragma unroll
        for (int r = 0; r < 8; ++r) {
            int i  = i0 + r + mo;
            int dc = nt * 16 + mr;
            ctxout[((size_t)b * TT + i) * DD + h * DH + dc] = f2bf(o[nt][r] * rinv[r]);
        }
}

// ---------------------------------------------------------------------------
// Kernel 4: output projection + bias + residual -> f32 scratch y
// ---------------------------------------------------------------------------
__global__ __launch_bounds__(128) void gemm_out_kernel(
    const u16* __restrict__ ctxb, const u32* __restrict__ wof,
    const float* __restrict__ x, const float* __restrict__ bias,
    float* __restrict__ y)
{
    int tid = threadIdx.x, lane = tid & 31, wave = tid >> 5;
    int m0 = blockIdx.x * 128 + (wave & 1) * 64;
    int n0 = blockIdx.y * 128 + (wave >> 1) * 64;
    int mr = lane & 15, hs = lane >> 4;

    v8f acc[4][4] = {};
    for (int kc = 0; kc < 32; ++kc) {
        BF16Frag a[4], b[4];
#pragma unroll
        for (int mt = 0; mt < 4; ++mt) {
            const u16* p = ctxb + (size_t)(m0 + mt * 16 + mr) * DD + kc * 32 + hs * 16;
            a[mt].u[0] = *(const u32x4*)p;
            a[mt].u[1] = *(const u32x4*)(p + 8);
        }
#pragma unroll
        for (int nt = 0; nt < 4; ++nt) {
            const u32* p = wof + ((size_t)(kc * 64 + (n0 >> 4) + nt) * 32 + lane) * 8;
            b[nt].u[0] = *(const u32x4*)p;
            b[nt].u[1] = *(const u32x4*)(p + 4);
        }
#pragma unroll
        for (int mt = 0; mt < 4; ++mt)
#pragma unroll
            for (int nt = 0; nt < 4; ++nt)
                acc[mt][nt] = wmma_bf16(a[mt].v, b[nt].v, acc[mt][nt]);
    }
#pragma unroll
    for (int mt = 0; mt < 4; ++mt)
        for (int nt = 0; nt < 4; ++nt)
#pragma unroll
            for (int r = 0; r < 8; ++r) {
                int grow = m0 + mt * 16 + r + hs * 8;
                int gcol = n0 + nt * 16 + mr;
                size_t off = (size_t)grow * DD + gcol;
                y[off] = x[off] + bias[gcol] + acc[mt][nt][r];
            }
}

// ---------------------------------------------------------------------------
// Kernel 5: LayerNorm per row of y -> d_out
// ---------------------------------------------------------------------------
__global__ __launch_bounds__(256) void ln_kernel(
    const float* __restrict__ y, const float* __restrict__ gamma,
    const float* __restrict__ beta, float* __restrict__ out)
{
    __shared__ float red[256];
    int row = blockIdx.x, tid = threadIdx.x;
    const float* yr = y + (size_t)row * DD;

    float v[4]; float s = 0.f;
#pragma unroll
    for (int i = 0; i < 4; ++i) { v[i] = yr[tid + i * 256]; s += v[i]; }
    red[tid] = s; __syncthreads();
    for (int o = 128; o > 0; o >>= 1) { if (tid < o) red[tid] += red[tid + o]; __syncthreads(); }
    float mu = red[0] * (1.0f / DD);
    __syncthreads();

    float s2 = 0.f;
#pragma unroll
    for (int i = 0; i < 4; ++i) { float d = v[i] - mu; s2 += d * d; }
    red[tid] = s2; __syncthreads();
    for (int o = 128; o > 0; o >>= 1) { if (tid < o) red[tid] += red[tid + o]; __syncthreads(); }
    float inv = rsqrtf(red[0] * (1.0f / DD) + 1e-5f);

#pragma unroll
    for (int i = 0; i < 4; ++i) {
        int col = tid + i * 256;
        out[(size_t)row * DD + col] = (v[i] - mu) * inv * gamma[col] + beta[col];
    }
}

// ---------------------------------------------------------------------------
extern "C" void kernel_launch(void* const* d_in, const int* in_sizes, int n_in,
                              void* d_out, int out_size, void* d_ws, size_t ws_size,
                              hipStream_t stream) {
    (void)in_sizes; (void)n_in; (void)out_size; (void)ws_size;
    const float* x    = (const float*)d_in[0];
    const float* cx   = (const float*)d_in[1];
    const float* lut  = (const float*)d_in[2];
    const float* Wq   = (const float*)d_in[3];
    const float* Wk   = (const float*)d_in[4];
    const float* Wv   = (const float*)d_in[5];
    const float* Wo   = (const float*)d_in[6];
    const float* bo   = (const float*)d_in[7];
    const float* gam  = (const float*)d_in[8];
    const float* bet  = (const float*)d_in[9];

    char* ws = (char*)d_ws;
    const size_t MB = 1ull << 20;
    u16* xb  = (u16*)(ws + 0 * MB);     // 4 MB
    u16* cb  = (u16*)(ws + 4 * MB);     // 4 MB
    u32* wqf = (u32*)(ws + 8 * MB);     // 2 MB
    u32* wkf = (u32*)(ws + 10 * MB);    // 2 MB
    u32* wvf = (u32*)(ws + 12 * MB);    // 2 MB
    u32* wof = (u32*)(ws + 14 * MB);    // 2 MB
    u32* ef  = (u32*)(ws + 16 * MB);    // 10 KB
    u16* Qw  = (u16*)(ws + 17 * MB);    // 4 MB
    u16* Kw  = (u16*)(ws + 21 * MB);    // 4 MB
    u16* Vt  = (u16*)(ws + 25 * MB);    // 4 MB
    u16* cxw = (u16*)(ws + 29 * MB);    // 4 MB
    float* yw = (float*)(ws + 33 * MB); // 8 MB  (total 41 MB)

    pack_kernel<<<dim3((PACK_TOTAL + 255) / 256), dim3(256), 0, stream>>>(
        x, cx, lut, Wq, Wk, Wv, Wo, xb, cb, wqf, wkf, wvf, wof, ef);

    gemm_qkv_kernel<<<dim3(16, 8, 3), dim3(128), 0, stream>>>(
        xb, cb, wqf, wkf, wvf, Qw, Kw, Vt);

    attn_kernel<<<dim3(512), dim3(128), 0, stream>>>(Qw, Kw, Vt, ef, cxw);

    gemm_out_kernel<<<dim3(16, 8), dim3(128), 0, stream>>>(cxw, wof, x, bo, yw);

    ln_kernel<<<dim3(MROWS), dim3(256), 0, stream>>>(yw, gam, bet, (float*)d_out);
}
